// GEOFeatureExtractor_54202487276059
// MI455X (gfx1250) — matrix operand order repaired and checked by reading, not compiled, not run on hardware
//
#include <hip/hip_runtime.h>
#include <hip/hip_bf16.h>

#define P_TOT   40000   // H*W pixels
#define C_DIM   256
#define R_DIM   1024
#define N_NODES 300
#define N_PAD   320     // padded to 5 tiles of 64

typedef __bf16 bf16_t;
typedef __attribute__((ext_vector_type(16))) __bf16 v16bf;
typedef __attribute__((ext_vector_type(8)))  __bf16 v8bf;
typedef __attribute__((ext_vector_type(8)))  float  v8f;

// A-matrix 16x32 fill order (matches B): lanes 0-15 hold K={0..15},
// lanes 16-31 hold K={16..31} for row M = lane%16.
__device__ __forceinline__ int a_k_idx(int e, int hf) {
  return hf * 16 + e;
}

// Async global->LDS copy of 16 bytes (gfx1250), tracked by ASYNCcnt.
__device__ __forceinline__ void async_copy_b128(unsigned lds_off, const void* gaddr) {
  asm volatile("global_load_async_to_lds_b128 %0, %1, off"
               :: "v"(lds_off), "v"((unsigned long long)(uintptr_t)gaddr)
               : "memory");
}
__device__ __forceinline__ void wait_async0() {
  asm volatile("s_wait_asynccnt 0" ::: "memory");
}

// ---------------------------------------------------------------------------
// Stage 0: transpose byte mask [p][n] -> bf16 mask [n][p] (padded to 320 rows)
// 64x64 tiles through LDS so both global sides are lane-contiguous.
// ---------------------------------------------------------------------------
__global__ void __launch_bounds__(256)
k_mask_T(const unsigned char* __restrict__ masks, unsigned short* __restrict__ Mraw) {
  bf16_t* Mbf = reinterpret_cast<bf16_t*>(Mraw);
  const int p0 = blockIdx.x * 64;
  const int n0 = blockIdx.y * 64;
  __shared__ unsigned char tile[64][65];

  const int t = threadIdx.x;
  const int c = t & 63;         // lane-contiguous index
  const int rb = t >> 6;        // 0..3
#pragma unroll
  for (int ii = 0; ii < 16; ++ii) {
    int i = rb + ii * 4;        // pixel row within tile
    int n = n0 + c;
    tile[i][c] = (n < N_NODES) ? masks[(size_t)(p0 + i) * N_NODES + n] : (unsigned char)0;
  }
  __syncthreads();
#pragma unroll
  for (int jj = 0; jj < 16; ++jj) {
    int j = rb + jj * 4;        // node row within tile
    Mbf[(size_t)(n0 + j) * P_TOT + (p0 + c)] = (bf16_t)(float)tile[c][j];
  }
}

// ---------------------------------------------------------------------------
// counts[n] = sum_p mask[n, p] from the transposed bf16 mask (contiguous rows)
// ---------------------------------------------------------------------------
__global__ void __launch_bounds__(256)
k_counts(const unsigned short* __restrict__ Mraw, float* __restrict__ counts) {
  const bf16_t* Mbf = reinterpret_cast<const bf16_t*>(Mraw);
  __shared__ float red[256];
  int n = blockIdx.x;
  float s = 0.f;
  for (int p = threadIdx.x; p < P_TOT; p += 256)
    s += (float)Mbf[(size_t)n * P_TOT + p];
  red[threadIdx.x] = s;
  __syncthreads();
  for (int off = 128; off > 0; off >>= 1) {
    if (threadIdx.x < off) red[threadIdx.x] += red[threadIdx.x + off];
    __syncthreads();
  }
  if (threadIdx.x == 0) counts[n] = red[0];
}

// ---------------------------------------------------------------------------
// Stage 1: H_k[r, p] = relu( sum_c x[c,p] * Wk1[c,r] + bk1[r] ), stored bf16
// TRANSPOSED [r][p] so stage-2 can stream it with async b128 copies.
// Block = 128 threads (4 waves), 64(p) x 64(r) tile per block.
// ---------------------------------------------------------------------------
__global__ void __launch_bounds__(128)
k_mlp1(const float* __restrict__ x,
       const float* __restrict__ W1p, const float* __restrict__ W1l, const float* __restrict__ W1c,
       const float* __restrict__ b1p, const float* __restrict__ b1l, const float* __restrict__ b1c,
       unsigned short* __restrict__ Hraw) {
  const int kb = blockIdx.z;
  const float* W1 = (kb == 0) ? W1p : (kb == 1) ? W1l : W1c;
  const float* b1 = (kb == 0) ? b1p : (kb == 1) ? b1l : b1c;
  const int p0 = blockIdx.y * 64;
  const int r0 = blockIdx.x * 64;
  bf16_t* Hbuf = reinterpret_cast<bf16_t*>(Hraw);

  __shared__ alignas(16) bf16_t tA[64][40]; // [m = pixel][k = channel]
  __shared__ alignas(16) bf16_t tB[64][40]; // [n = r-col][k = channel]

  const int t = threadIdx.x;
  const int lane = t & 31, w = t >> 5;
  const int mrow = lane & 15, hf = lane >> 4;
  const int mbase = w * 16;

  v8f acc[4] = {};
  for (int c0 = 0; c0 < C_DIM; c0 += 32) {
    __syncthreads();
#pragma unroll
    for (int i = 0; i < 16; ++i) {
      int idx = t + i * 128;       // 2048 elements per tile, lane-coalesced
      int k = idx >> 6, m = idx & 63;
      tA[m][k] = (bf16_t)x[(size_t)(c0 + k) * P_TOT + (p0 + m)];
      tB[m][k] = (bf16_t)W1[(size_t)(c0 + k) * R_DIM + (r0 + m)];
    }
    __syncthreads();
    v16bf a;
#pragma unroll
    for (int e = 0; e < 16; ++e) a[e] = tA[mbase + mrow][a_k_idx(e, hf)];
#pragma unroll
    for (int ch = 0; ch < 4; ++ch) {
      v16bf b;
#pragma unroll
      for (int e = 0; e < 16; ++e) b[e] = tB[ch * 16 + mrow][hf * 16 + e];
      acc[ch] = __builtin_amdgcn_wmma_f32_16x16x32_bf16(
          false, a, false, b, (short)0, acc[ch], false, false);
    }
  }
  // Epilogue: bias + ReLU; C/D layout: VGPR j -> M(pixel) = j + hf*8, N(r) = lane%16.
  // With H stored [r][p], each lane writes 8 contiguous bf16 -> one b128 store.
#pragma unroll
  for (int ch = 0; ch < 4; ++ch) {
    int r = r0 + ch * 16 + mrow;
    float bias = b1[r];
    v8bf hv;
#pragma unroll
    for (int j = 0; j < 8; ++j) {
      float v = acc[ch][j] + bias;
      hv[j] = (bf16_t)(v > 0.f ? v : 0.f);
    }
    int p = p0 + mbase + hf * 8;
    *reinterpret_cast<v8bf*>(&Hbuf[((size_t)kb * R_DIM + r) * P_TOT + p]) = hv;
  }
}

// ---------------------------------------------------------------------------
// Stage 2: A_k = mask^T @ H_k  -> Abuf[kb][node][r] (f32).  K = 40000.
// Both operands are pre-laid-out row-contiguous bf16, so tiles are streamed
// straight into LDS with global_load_async_to_lds_b128 (no VGPR staging).
// ---------------------------------------------------------------------------
__global__ void __launch_bounds__(128)
k_masked_sum(const unsigned short* __restrict__ Mraw,
             const unsigned short* __restrict__ Hraw,
             float* __restrict__ Abuf) {
  const int kb = blockIdx.z;
  const int n0 = blockIdx.y * 64;  // node tile
  const int r0 = blockIdx.x * 64;
  const bf16_t* Mbf = reinterpret_cast<const bf16_t*>(Mraw);
  const bf16_t* Hk  = reinterpret_cast<const bf16_t*>(Hraw) + (size_t)kb * R_DIM * P_TOT;

  __shared__ alignas(16) bf16_t tA[64][40]; // [m = node][k = pixel], rows 80B (16B-aligned)
  __shared__ alignas(16) bf16_t tB[64][40]; // [n = r  ][k = pixel]

  const int t = threadIdx.x;
  const int lane = t & 31, w = t >> 5;
  const int mrow = lane & 15, hf = lane >> 4;
  const int mbase = w * 16;

  // Per-thread async-copy assignment: 2 x (row, 16B segment) per tile.
  // seg id = t + i*128; row = sid>>2 (0..63), seg = (sid&3)*8 elements.
  v8f acc[4] = {};
  for (int p0 = 0; p0 < P_TOT; p0 += 32) {
    __syncthreads();  // prior iteration's fragment reads are complete (waited before WMMA)
#pragma unroll
    for (int i = 0; i < 2; ++i) {
      int sid = t + i * 128;
      int row = sid >> 2, sg = (sid & 3) * 8;
      async_copy_b128((unsigned)(uintptr_t)&tA[row][sg],
                      Mbf + (size_t)(n0 + row) * P_TOT + p0 + sg);
      async_copy_b128((unsigned)(uintptr_t)&tB[row][sg],
                      Hk + (size_t)(r0 + row) * P_TOT + p0 + sg);
    }
    wait_async0();
    __syncthreads();
    v16bf a;
#pragma unroll
    for (int e = 0; e < 16; ++e) a[e] = tA[mbase + mrow][a_k_idx(e, hf)];
#pragma unroll
    for (int ch = 0; ch < 4; ++ch) {
      v16bf b;
#pragma unroll
      for (int e = 0; e < 16; ++e) b[e] = tB[ch * 16 + mrow][hf * 16 + e];
      acc[ch] = __builtin_amdgcn_wmma_f32_16x16x32_bf16(
          false, a, false, b, (short)0, acc[ch], false, false);
    }
  }
#pragma unroll
  for (int ch = 0; ch < 4; ++ch) {
    int r = r0 + ch * 16 + mrow;
#pragma unroll
    for (int j = 0; j < 8; ++j) {
      int node = n0 + mbase + hf * 8 + j;
      Abuf[((size_t)kb * N_PAD + node) * R_DIM + r] = acc[ch][j];
    }
  }
}

// ---------------------------------------------------------------------------
// Stage 3: out[n,:] = (A_{label[n]}[n,:]/count[n]) @ W2_{label[n]} + b2_{label[n]}
// Computes (A_k/count) @ Wk2 for all k; writes only rows whose label == k.
// ---------------------------------------------------------------------------
__global__ void __launch_bounds__(128)
k_mlp2_select(const float* __restrict__ Abuf,
              const float* __restrict__ counts,
              const float* __restrict__ W2p, const float* __restrict__ W2l, const float* __restrict__ W2c,
              const float* __restrict__ b2p, const float* __restrict__ b2l, const float* __restrict__ b2c,
              const int* __restrict__ labels,
              float* __restrict__ out) {
  const int kb = blockIdx.z;
  const float* W2 = (kb == 0) ? W2p : (kb == 1) ? W2l : W2c;
  const float* b2 = (kb == 0) ? b2p : (kb == 1) ? b2l : b2c;
  const int n0 = blockIdx.y * 64;
  const int r0 = blockIdx.x * 64;

  __shared__ alignas(16) bf16_t tA[64][40]; // [m = node][k]  (pre-scaled by 1/count)
  __shared__ alignas(16) bf16_t tB[64][40]; // [n = r   ][k]
  __shared__ float inv[64];

  const int t = threadIdx.x;
  const int lane = t & 31, w = t >> 5;
  const int mrow = lane & 15, hf = lane >> 4;
  const int mbase = w * 16;

  if (t < 64) {
    float c = counts[n0 + t];
    inv[t] = (c > 0.f) ? 1.f / c : 0.f;
  }
  __syncthreads();

  v8f acc[4] = {};
  for (int k0 = 0; k0 < R_DIM; k0 += 32) {
    __syncthreads();
#pragma unroll
    for (int i = 0; i < 16; ++i) {
      int idx = t + i * 128;
      // A: k fastest (row of 32 floats per node is contiguous)
      int ka = idx & 31, ma = idx >> 5;
      tA[ma][ka] = (bf16_t)(Abuf[((size_t)kb * N_PAD + n0 + ma) * R_DIM + (k0 + ka)] * inv[ma]);
      // B: n fastest (row of W2 is contiguous in r)
      int kbi = idx >> 6, mb = idx & 63;
      tB[mb][kbi] = (bf16_t)W2[(size_t)(k0 + kbi) * R_DIM + (r0 + mb)];
    }
    __syncthreads();
    v16bf a;
#pragma unroll
    for (int e = 0; e < 16; ++e) a[e] = tA[mbase + mrow][a_k_idx(e, hf)];
#pragma unroll
    for (int ch = 0; ch < 4; ++ch) {
      v16bf b;
#pragma unroll
      for (int e = 0; e < 16; ++e) b[e] = tB[ch * 16 + mrow][hf * 16 + e];
      acc[ch] = __builtin_amdgcn_wmma_f32_16x16x32_bf16(
          false, a, false, b, (short)0, acc[ch], false, false);
    }
  }
#pragma unroll
  for (int ch = 0; ch < 4; ++ch) {
    int r = r0 + ch * 16 + mrow;
    float bias = b2[r];
#pragma unroll
    for (int j = 0; j < 8; ++j) {
      int node = n0 + mbase + hf * 8 + j;
      if (node < N_NODES && labels[node] == kb)
        out[(size_t)node * R_DIM + r] = acc[ch][j] + bias;
    }
  }
}

// ---------------------------------------------------------------------------
extern "C" void kernel_launch(void* const* d_in, const int* in_sizes, int n_in,
                              void* d_out, int out_size, void* d_ws, size_t ws_size,
                              hipStream_t stream) {
  const float*         x      = (const float*)d_in[0];
  const unsigned char* masks  = (const unsigned char*)d_in[1]; // bool -> 1 byte
  const int*           labels = (const int*)d_in[2];
  const float* Wp1 = (const float*)d_in[3];
  const float* bp1 = (const float*)d_in[4];
  const float* Wp2 = (const float*)d_in[5];
  const float* bp2 = (const float*)d_in[6];
  const float* Wl1 = (const float*)d_in[7];
  const float* bl1 = (const float*)d_in[8];
  const float* Wl2 = (const float*)d_in[9];
  const float* bl2 = (const float*)d_in[10];
  const float* Wc1 = (const float*)d_in[11];
  const float* bc1 = (const float*)d_in[12];
  const float* Wc2 = (const float*)d_in[13];
  const float* bc2 = (const float*)d_in[14];
  float* out = (float*)d_out;

  // Workspace layout (all 16B-aligned chunks)
  const size_t H_BYTES = (size_t)3 * R_DIM * P_TOT * sizeof(unsigned short); // 245,760,000
  const size_t M_BYTES = (size_t)N_PAD * P_TOT * sizeof(unsigned short);     //  25,600,000
  const size_t A_BYTES = (size_t)3 * N_PAD * R_DIM * sizeof(float);          //   3,932,160
  char* wsb = (char*)d_ws;
  unsigned short* Hbuf   = (unsigned short*)wsb;
  unsigned short* Mbf    = (unsigned short*)(wsb + H_BYTES);
  float*          Abuf   = (float*)(wsb + H_BYTES + M_BYTES);
  float*          counts = (float*)(wsb + H_BYTES + M_BYTES + A_BYTES);

  // Stage 0: mask transpose to bf16 [320][40000]
  k_mask_T<<<dim3(P_TOT / 64, N_PAD / 64), 256, 0, stream>>>(masks, Mbf);

  // counts[n] from transposed mask (contiguous rows)
  k_counts<<<dim3(N_PAD), 256, 0, stream>>>(Mbf, counts);

  // Stage 1: H_k = relu(x^T Wk1 + bk1), stored [kb][r][p] bf16
  k_mlp1<<<dim3(R_DIM / 64, P_TOT / 64, 3), 128, 0, stream>>>(
      x, Wp1, Wl1, Wc1, bp1, bl1, bc1, Hbuf);

  // Stage 2: A_k = mask^T H_k   [3 x 320 x 1024] f32 (async-LDS streamed)
  k_masked_sum<<<dim3(R_DIM / 64, N_PAD / 64, 3), 128, 0, stream>>>(
      Mbf, Hbuf, Abuf);

  // Stage 3: (A_k / count) @ Wk2 + bk2, gathered by label into out [300 x 1024]
  k_mlp2_select<<<dim3(R_DIM / 64, N_PAD / 64, 3), 128, 0, stream>>>(
      Abuf, counts, Wp2, Wl2, Wc2, bp2, bl2, bc2, labels, out);
}
